// AtomSelector_16595753631976
// MI455X (gfx1250) — matrix operand order repaired and checked by reading, not compiled, not run on hardware
//
#include <hip/hip_runtime.h>
#include <hip/hip_bf16.h>

// ---------------------------------------------------------------------------
// AtomSelector on MI455X (gfx1250): f16 WMMA GEMMs with register blocking,
// batched tile loads + pointer-strided K loop, fused GRU gates, one-hot
// gumbel-argmax via packed u64 atomicMax.
// ---------------------------------------------------------------------------

typedef __attribute__((ext_vector_type(16))) _Float16 v16h;
typedef __attribute__((ext_vector_type(8)))  _Float16 v8h;
typedef __attribute__((ext_vector_type(8)))  float    v8f;

static constexpr int Bsz = 4096;  // batch
static constexpr int Asz = 4096;  // atoms
static constexpr int Hd  = 768;   // hidden
static constexpr int Ls  = 4;     // steps

// ---- WMMA wrapper (codegen-confirmed builtin, 8 args) ----------------------
__device__ __forceinline__ v8f wmma_f16(v16h a, v16h b, v8f c) {
    return __builtin_amdgcn_wmma_f32_16x16x32_f16(
        /*neg_a=*/false, a, /*neg_b=*/false, b,
        /*c_mod=*/(short)0, c, /*reuse_a=*/false, /*reuse_b=*/false);
}

// ---- per-lane tile pointers (hoisted out of K loop, advanced by +32) -------
// A-matrix 16x32 f16 (ISA 7.12.2): lanes 0-15: M=lane, chunks K[+0..7],[+16..23]
//                                  lanes 16-31: M=lane-16, chunks [+8..15],[+24..31]
__device__ __forceinline__ const _Float16* a_tile_ptr(const _Float16* base, int ldk,
                                                      int m0, int lane) {
    return base + (size_t)(m0 + (lane & 15)) * ldk + ((lane >> 4) << 3);
}
__device__ __forceinline__ v16h load_a_from(const _Float16* p) {
    v8h lo = *(const v8h*)(p);
    v8h hi = *(const v8h*)(p + 16);
    return __builtin_shufflevector(lo, hi, 0,1,2,3,4,5,6,7,8,9,10,11,12,13,14,15);
}
// B-matrix 32x16: B[k,n] = W[n,k], W row-major [N x K]; lane holds column
// n=lane&15, chunks K[+0..7],[+16..23] (lanes 16-31: [+8..15],[+24..31]).
__device__ __forceinline__ const _Float16* b_tile_ptr(const _Float16* w, int ldk,
                                                      int n0, int lane) {
    return w + (size_t)(n0 + (lane & 15)) * ldk + ((lane >> 4) << 3);
}
__device__ __forceinline__ v16h load_b_from(const _Float16* p) {
    v8h lo = *(const v8h*)(p);
    v8h hi = *(const v8h*)(p + 16);
    return __builtin_shufflevector(lo, hi, 0,1,2,3,4,5,6,7,8,9,10,11,12,13,14,15);
}

// ---- pseudo-gumbel (PCG hash; JAX bits are unreproducible anyway) ----------
__device__ __forceinline__ unsigned pcg_hash(unsigned x) {
    x = x * 747796405u + 2891336453u;
    unsigned w = ((x >> ((x >> 28u) + 4u)) ^ x) * 277803737u;
    return (w >> 22u) ^ w;
}
__device__ __forceinline__ float gumbelf(unsigned seed) {
    float u = ((float)(pcg_hash(seed) >> 8) + 0.5f) * (1.0f / 16777216.0f);
    return -__logf(-__logf(u));
}

// ===========================================================================
// init: f32->f16 weight/activation conversion, zero h, zero packed argmax
// ===========================================================================
__global__ __launch_bounds__(256)
void convert_kernel(const float* __restrict__ cls, const float* __restrict__ wih,
                    const float* __restrict__ whh, const float* __restrict__ hw,
                    _Float16* __restrict__ cls16, _Float16* __restrict__ wih16,
                    _Float16* __restrict__ whh16, _Float16* __restrict__ hw16,
                    float* __restrict__ h32, _Float16* __restrict__ h16,
                    unsigned long long* __restrict__ packed) {
    size_t i = (size_t)blockIdx.x * blockDim.x + threadIdx.x;
    size_t stride = (size_t)gridDim.x * blockDim.x;
    for (size_t k = i; k < (size_t)Bsz * Hd; k += stride) {
        cls16[k] = (_Float16)cls[k];
        h32[k] = 0.0f;
        h16[k] = (_Float16)0.0f;
    }
    for (size_t k = i; k < (size_t)3 * Hd * Hd; k += stride) {
        wih16[k] = (_Float16)wih[k];
        whh16[k] = (_Float16)whh[k];
    }
    for (size_t k = i; k < (size_t)Asz * Hd; k += stride) {
        hw16[k] = (_Float16)hw[k];
    }
    for (size_t k = i; k < (size_t)Bsz; k += stride) packed[k] = 0ull;
}

// mask init: mask = x_ (0/1 bytes); empty rows get col0 = 1 (step-0 rule)
__global__ __launch_bounds__(256)
void mask_init_kernel(const float* __restrict__ x_, unsigned char* __restrict__ mask) {
    __shared__ float red[256];
    int b = blockIdx.x, t = threadIdx.x;
    float s = 0.0f;
    for (int a = t; a < Asz; a += 256) {
        float v = x_[(size_t)b * Asz + a];
        mask[(size_t)b * Asz + a] = (v > 0.5f) ? 1 : 0;
        s += v;
    }
    red[t] = s; __syncthreads();
    for (int o = 128; o > 0; o >>= 1) { if (t < o) red[t] += red[t + o]; __syncthreads(); }
    if (t == 0 && red[0] == 0.0f) mask[(size_t)b * Asz] = 1;
}

// ===========================================================================
// GRU step: fused gi/gh GEMMs, register-blocked 32 rows x 16 gate-cols.
// All 10 tiles of a K-step loaded before the 12 WMMAs (one wait, then burst).
// ===========================================================================
__global__ __launch_bounds__(256)
void gru_kernel(const _Float16* __restrict__ xin,    // cur_input f16 [B,H]
                const _Float16* __restrict__ hin16,  // h f16 [B,H]
                const float*    __restrict__ hin32,  // h f32 [B,H]
                const _Float16* __restrict__ wih16,  // [3H,H]
                const _Float16* __restrict__ whh16,  // [3H,H]
                const float* __restrict__ bih, const float* __restrict__ bhh,
                float* __restrict__ hout32, _Float16* __restrict__ hout16) {
    const int lane = threadIdx.x & 31;
    const int wave = threadIdx.x >> 5;
    const int tile = blockIdx.x * 8 + wave;
    const int CT = Hd / 16;                    // 48 col-tiles
    const int m0 = (tile / CT) * 32;           // 32-row macro tile
    const int c0 = (tile % CT) * 16;

    // per-lane tile pointers, advanced by 32 elements each K-step
    const _Float16* pax0 = a_tile_ptr(xin,   Hd, m0,      lane);
    const _Float16* pax1 = a_tile_ptr(xin,   Hd, m0 + 16, lane);
    const _Float16* pah0 = a_tile_ptr(hin16, Hd, m0,      lane);
    const _Float16* pah1 = a_tile_ptr(hin16, Hd, m0 + 16, lane);
    const _Float16* pb0  = b_tile_ptr(wih16, Hd, c0,          lane);
    const _Float16* pb1  = b_tile_ptr(wih16, Hd, c0 + Hd,     lane);
    const _Float16* pb2  = b_tile_ptr(wih16, Hd, c0 + 2 * Hd, lane);
    const _Float16* pb3  = b_tile_ptr(whh16, Hd, c0,          lane);
    const _Float16* pb4  = b_tile_ptr(whh16, Hd, c0 + Hd,     lane);
    const _Float16* pb5  = b_tile_ptr(whh16, Hd, c0 + 2 * Hd, lane);

    v8f air[2] = {v8f{}, v8f{}}, aiz[2] = {v8f{}, v8f{}}, ain[2] = {v8f{}, v8f{}};
    v8f ahr[2] = {v8f{}, v8f{}}, ahz[2] = {v8f{}, v8f{}}, ahn[2] = {v8f{}, v8f{}};

    for (int k0 = 0; k0 < Hd; k0 += 32) {
        if (k0 + 32 < Hd) {  // global_prefetch_b8 next K-slab
            __builtin_prefetch(pax0 + 32, 0, 1);
            __builtin_prefetch(pah0 + 32, 0, 1);
        }
        // batch ALL tile loads first (single clause burst, one wait point)
        v16h ax0 = load_a_from(pax0);
        v16h ax1 = load_a_from(pax1);
        v16h ah0 = load_a_from(pah0);
        v16h ah1 = load_a_from(pah1);
        v16h b0  = load_b_from(pb0);
        v16h b1  = load_b_from(pb1);
        v16h b2  = load_b_from(pb2);
        v16h b3  = load_b_from(pb3);
        v16h b4  = load_b_from(pb4);
        v16h b5  = load_b_from(pb5);

        air[0] = wmma_f16(ax0, b0, air[0]);  air[1] = wmma_f16(ax1, b0, air[1]);
        aiz[0] = wmma_f16(ax0, b1, aiz[0]);  aiz[1] = wmma_f16(ax1, b1, aiz[1]);
        ain[0] = wmma_f16(ax0, b2, ain[0]);  ain[1] = wmma_f16(ax1, b2, ain[1]);
        ahr[0] = wmma_f16(ah0, b3, ahr[0]);  ahr[1] = wmma_f16(ah1, b3, ahr[1]);
        ahz[0] = wmma_f16(ah0, b4, ahz[0]);  ahz[1] = wmma_f16(ah1, b4, ahz[1]);
        ahn[0] = wmma_f16(ah0, b5, ahn[0]);  ahn[1] = wmma_f16(ah1, b5, ahn[1]);

        pax0 += 32; pax1 += 32; pah0 += 32; pah1 += 32;
        pb0 += 32; pb1 += 32; pb2 += 32; pb3 += 32; pb4 += 32; pb5 += 32;
    }

    // epilogue: C layout — VGPR v: lanes 0-15 => M=v, lanes 16-31 => M=v+8
    const int n = lane & 15;
    const int c = c0 + n;
    const float vbir = bih[c], vbiz = bih[Hd + c], vbin = bih[2 * Hd + c];
    const float vbhr = bhh[c], vbhz = bhh[Hd + c], vbhn = bhh[2 * Hd + c];
    #pragma unroll
    for (int i = 0; i < 2; ++i) {
        const int mbase = m0 + 16 * i + ((lane >> 4) << 3);
        #pragma unroll
        for (int v = 0; v < 8; ++v) {
            const int m = mbase + v;
            float r = 1.0f / (1.0f + __expf(-((air[i][v] + vbir) + (ahr[i][v] + vbhr))));
            float z = 1.0f / (1.0f + __expf(-((aiz[i][v] + vbiz) + (ahz[i][v] + vbhz))));
            float t = (ain[i][v] + vbin) + r * (ahn[i][v] + vbhn);
            float nn = 1.0f - 2.0f / (__expf(2.0f * t) + 1.0f);   // tanh
            float hold = hin32[(size_t)m * Hd + c];
            float hnew = (1.0f - z) * nn + z * hold;
            hout32[(size_t)m * Hd + c] = hnew;
            hout16[(size_t)m * Hd + c] = (_Float16)hnew;
        }
    }
}

// ===========================================================================
// head: out = h @ head_w.T + b, register-blocked 32x64 macro-tile per wave
// (2 m-tiles x 4 a-tiles). Batched tile loads per K-step. Masked gumbel
// perturbation + per-row argmax: in-register column-tile max, 16-lane
// shuffle reduce, packed u64 atomicMax.
// ===========================================================================
__global__ __launch_bounds__(256)
void head_kernel(const _Float16* __restrict__ h16, const _Float16* __restrict__ hw16,
                 const float* __restrict__ hb, const unsigned char* __restrict__ mask,
                 unsigned long long* __restrict__ packed, int step) {
    const int lane = threadIdx.x & 31;
    const int wave = threadIdx.x >> 5;
    const int tile = blockIdx.x * 8 + wave;
    const int AT = Asz / 64;                   // 64 macro col-tiles
    const int m0 = (tile / AT) * 32;
    const int a0 = (tile % AT) * 64;

    const _Float16* pa0 = a_tile_ptr(h16, Hd, m0,      lane);
    const _Float16* pa1 = a_tile_ptr(h16, Hd, m0 + 16, lane);
    const _Float16* pb0 = b_tile_ptr(hw16, Hd, a0,      lane);
    const _Float16* pb1 = b_tile_ptr(hw16, Hd, a0 + 16, lane);
    const _Float16* pb2 = b_tile_ptr(hw16, Hd, a0 + 32, lane);
    const _Float16* pb3 = b_tile_ptr(hw16, Hd, a0 + 48, lane);

    v8f acc[2][4];
    #pragma unroll
    for (int i = 0; i < 2; ++i)
        #pragma unroll
        for (int j = 0; j < 4; ++j) acc[i][j] = v8f{};

    for (int k0 = 0; k0 < Hd; k0 += 32) {
        if (k0 + 32 < Hd) {
            __builtin_prefetch(pa0 + 32, 0, 1);
            __builtin_prefetch(pb0 + 32, 0, 1);
        }
        v16h a0t = load_a_from(pa0);
        v16h a1t = load_a_from(pa1);
        v16h b0  = load_b_from(pb0);
        v16h b1  = load_b_from(pb1);
        v16h b2  = load_b_from(pb2);
        v16h b3  = load_b_from(pb3);

        acc[0][0] = wmma_f16(a0t, b0, acc[0][0]);
        acc[1][0] = wmma_f16(a1t, b0, acc[1][0]);
        acc[0][1] = wmma_f16(a0t, b1, acc[0][1]);
        acc[1][1] = wmma_f16(a1t, b1, acc[1][1]);
        acc[0][2] = wmma_f16(a0t, b2, acc[0][2]);
        acc[1][2] = wmma_f16(a1t, b2, acc[1][2]);
        acc[0][3] = wmma_f16(a0t, b3, acc[0][3]);
        acc[1][3] = wmma_f16(a1t, b3, acc[1][3]);

        pa0 += 32; pa1 += 32; pb0 += 32; pb1 += 32; pb2 += 32; pb3 += 32;
    }

    const int n = lane & 15;
    #pragma unroll
    for (int i = 0; i < 2; ++i) {
        const int mbase = m0 + 16 * i + ((lane >> 4) << 3);
        #pragma unroll
        for (int v = 0; v < 8; ++v) {
            const int m = mbase + v;
            // local max over this lane's 4 column-tiles first (4x fewer shuffles)
            unsigned long long p = 0ull;
            #pragma unroll
            for (int j = 0; j < 4; ++j) {
                const int ag = a0 + 16 * j + n;
                if (mask[(size_t)m * Asz + ag]) {
                    unsigned seed = (unsigned)((size_t)m * Asz + ag)
                                  + (unsigned)step * 0x9E3779B9u;
                    float val = acc[i][j][v] + hb[ag] + gumbelf(seed);
                    unsigned ub = __float_as_uint(val);
                    ub = (ub & 0x80000000u) ? ~ub : (ub | 0x80000000u);
                    unsigned long long q = ((unsigned long long)ub << 32) | (unsigned)ag;
                    p = (p > q) ? p : q;
                }
            }
            // max over the 16 lanes holding this row (offs 1..8 stay in half)
            #pragma unroll
            for (int off = 1; off < 16; off <<= 1) {
                unsigned long long q = __shfl_xor(p, off, 32);
                p = (p > q) ? p : q;
            }
            if (n == 0) atomicMax(&packed[m], p);
        }
    }
}

// ===========================================================================
// select: one-hot output scatter, mask mutation for next step, next cur_input
// one block per row
// ===========================================================================
__global__ __launch_bounds__(256)
void select_kernel(unsigned long long* __restrict__ packed,
                   unsigned char* __restrict__ mask,
                   const float* __restrict__ cls, const float* __restrict__ aew,
                   _Float16* __restrict__ curin16, float* __restrict__ out, int step) {
    const int b = blockIdx.x;
    const int t = threadIdx.x;
    const unsigned long long p = packed[b];
    const unsigned idx = (unsigned)(p & 0xFFFFFFFFull);
    __syncthreads();                              // all threads read p before reset
    if (t == 0) {
        out[((size_t)b * Ls + step) * Asz + idx] = 1.0f;   // prob == y_hard numerically
        packed[b] = 0ull;                                  // re-arm for next step
    }
    if (step < Ls - 1) {
        // combined: end-of-step mask[idx]=0; next-step (idx==0 -> row=0); mask[0]=1
        for (int a = t; a < Asz; a += 256) {
            unsigned char mv = mask[(size_t)b * Asz + a];
            if (idx == 0u)            mv = (a == 0) ? 1 : 0;
            else {
                if (a == (int)idx)    mv = 0;
                if (a == 0)           mv = 1;
            }
            mask[(size_t)b * Asz + a] = mv;
        }
        // cur_input = cls + ae_weight[idx]  (one-hot matmul == row gather)
        for (int c = t; c < Hd; c += 256) {
            float xv = cls[(size_t)b * Hd + c] + aew[(size_t)idx * Hd + c];
            curin16[(size_t)b * Hd + c] = (_Float16)xv;
        }
    }
}

// ===========================================================================
extern "C" void kernel_launch(void* const* d_in, const int* in_sizes, int n_in,
                              void* d_out, int out_size, void* d_ws, size_t ws_size,
                              hipStream_t stream) {
    const float* cls = (const float*)d_in[0];
    const float* x_  = (const float*)d_in[1];
    const float* wih = (const float*)d_in[2];
    const float* whh = (const float*)d_in[3];
    const float* bih = (const float*)d_in[4];
    const float* bhh = (const float*)d_in[5];
    const float* hw  = (const float*)d_in[6];
    const float* hb  = (const float*)d_in[7];
    const float* aew = (const float*)d_in[8];
    float* out = (float*)d_out;

    // ---- workspace carve (256B aligned) ----
    char* ws = (char*)d_ws;
    size_t off = 0;
    auto carve = [&](size_t bytes) -> char* {
        char* p = ws + off;
        off += (bytes + 255) & ~(size_t)255;
        return p;
    };
    _Float16* wih16 = (_Float16*)carve((size_t)3 * Hd * Hd * 2);
    _Float16* whh16 = (_Float16*)carve((size_t)3 * Hd * Hd * 2);
    _Float16* hw16  = (_Float16*)carve((size_t)Asz * Hd * 2);
    _Float16* cls16 = (_Float16*)carve((size_t)Bsz * Hd * 2);
    _Float16* cur16 = (_Float16*)carve((size_t)Bsz * Hd * 2);
    _Float16* h16a  = (_Float16*)carve((size_t)Bsz * Hd * 2);
    _Float16* h16b  = (_Float16*)carve((size_t)Bsz * Hd * 2);
    float*    h32a  = (float*)   carve((size_t)Bsz * Hd * 4);
    float*    h32b  = (float*)   carve((size_t)Bsz * Hd * 4);
    unsigned char* mask = (unsigned char*)carve((size_t)Bsz * Asz);
    unsigned long long* packed = (unsigned long long*)carve((size_t)Bsz * 8);

    // output is one-hot rows: zero it first
    hipMemsetAsync(d_out, 0, (size_t)out_size * sizeof(float), stream);

    convert_kernel<<<2048, 256, 0, stream>>>(cls, wih, whh, hw, cls16, wih16,
                                             whh16, hw16, h32a, h16a, packed);
    mask_init_kernel<<<Bsz, 256, 0, stream>>>(x_, mask);

    const _Float16* cur = cls16;   // step 0: cur_input = cls
    for (int j = 0; j < Ls; ++j) {
        const _Float16* h16c = (j & 1) ? h16b : h16a;
        const float*    h32c = (j & 1) ? h32b : h32a;
        _Float16*       h16n = (j & 1) ? h16a : h16b;
        float*          h32n = (j & 1) ? h32a : h32b;

        // 32-row macro tiles: (Bsz/32)*(Hd/16) waves, 8 waves per block
        gru_kernel<<<(Bsz / 32) * (Hd / 16) / 8, 256, 0, stream>>>(
            cur, h16c, h32c, wih16, whh16, bih, bhh, h32n, h16n);
        // 32x64 macro tiles: (Bsz/32)*(Asz/64) waves, 8 waves per block
        head_kernel<<<(Bsz / 32) * (Asz / 64) / 8, 256, 0, stream>>>(
            h16n, hw16, hb, mask, packed, j);
        select_kernel<<<Bsz, 256, 0, stream>>>(packed, mask, cls, aew, cur16, out, j);
        cur = cur16;
    }
}
